// ESM2Backbone_15126874816553
// MI455X (gfx1250) — matrix operand order repaired and checked by reading, not compiled, not run on hardware
//
#include <hip/hip_runtime.h>
#include <math.h>

#define LSEQ   512
#define DMODEL 480
#define NHEAD  20
#define DHEAD  24
#define DHPAD  32
#define FDIM   1920
#define NLAYER 12
#define NLH    (NLAYER * NHEAD)   // 240
#define MASK_TOK 32
#define MASK_RATIO_TRAIN 0.12f
#define LN_EPS 1e-5f

typedef __attribute__((ext_vector_type(16))) __bf16 v16bf;
typedef __attribute__((ext_vector_type(8)))  __bf16 v8bf;
typedef __attribute__((ext_vector_type(8)))  float  v8f;

static __device__ __forceinline__ __bf16 f2bf(float f) {
    unsigned u = __float_as_uint(f);
    unsigned r = (u + 0x7FFFu + ((u >> 16) & 1u)) >> 16;
    unsigned short hs = (unsigned short)r;
    __bf16 b;
    __builtin_memcpy(&b, &hs, 2);
    return b;
}

// ---------------------------------------------------------------------------
// Embedding: token-dropout scale (single block) then scaled gather.
// ---------------------------------------------------------------------------
__global__ void esm_mask_stats(const int* __restrict__ ids,
                               const int* __restrict__ amask,
                               float* __restrict__ scale) {
    __shared__ float sm[512];
    __shared__ float ss[512];
    int t = threadIdx.x;
    sm[t] = (ids[t] == MASK_TOK) ? 1.f : 0.f;
    ss[t] = (float)amask[t];
    __syncthreads();
    for (int s = 256; s > 0; s >>= 1) {
        if (t < s) { sm[t] += sm[t + s]; ss[t] += ss[t + s]; }
        __syncthreads();
    }
    if (t == 0) {
        float ratio = sm[0] / ss[0];
        scale[0] = (1.0f - MASK_RATIO_TRAIN) / (1.0f - ratio);
    }
}

__global__ void esm_embed(const int* __restrict__ ids,
                          const int* __restrict__ amask,
                          const float* __restrict__ we,
                          const float* __restrict__ scale,
                          float* __restrict__ h) {
    int l = blockIdx.x;
    int id = ids[l];
    float s = scale[0] * (float)amask[l];
    for (int d = threadIdx.x; d < DMODEL; d += blockDim.x) {
        float v = (id == MASK_TOK) ? 0.f : we[(long)id * DMODEL + d];
        h[(long)l * DMODEL + d] = v * s;
    }
}

// ---------------------------------------------------------------------------
// LayerNorm: one block per row. Optional f32 and/or bf16 outputs.
// ---------------------------------------------------------------------------
__global__ void esm_layernorm(const float* __restrict__ x,
                              const float* __restrict__ w,
                              const float* __restrict__ b,
                              float* __restrict__ outF,
                              __bf16* __restrict__ outB) {
    __shared__ float s0[256], s1[256];
    int row = blockIdx.x;
    const float* xr = x + (long)row * DMODEL;
    float sum = 0.f, sq = 0.f;
    for (int d = threadIdx.x; d < DMODEL; d += blockDim.x) {
        float v = xr[d];
        sum += v; sq += v * v;
    }
    s0[threadIdx.x] = sum; s1[threadIdx.x] = sq;
    __syncthreads();
    for (int s = 128; s > 0; s >>= 1) {
        if (threadIdx.x < s) {
            s0[threadIdx.x] += s0[threadIdx.x + s];
            s1[threadIdx.x] += s1[threadIdx.x + s];
        }
        __syncthreads();
    }
    float mean = s0[0] * (1.0f / DMODEL);
    float var  = s1[0] * (1.0f / DMODEL) - mean * mean;
    float inv  = rsqrtf(var + LN_EPS);
    for (int d = threadIdx.x; d < DMODEL; d += blockDim.x) {
        float y = (xr[d] - mean) * inv * w[d] + b[d];
        if (outF) outF[(long)row * DMODEL + d] = y;
        if (outB) outB[(long)row * DMODEL + d] = f2bf(y);
    }
}

// ---------------------------------------------------------------------------
// Weight convert + transpose: f32 (K x N) -> bf16 (N x K)
// ---------------------------------------------------------------------------
__global__ void esm_w2bf_t(const float* __restrict__ src,
                           __bf16* __restrict__ dst, int K, int N) {
    long total = (long)K * N;
    for (long i = (long)blockIdx.x * blockDim.x + threadIdx.x; i < total;
         i += (long)gridDim.x * blockDim.x) {
        long n = i / K, k = i % K;
        dst[i] = f2bf(src[k * (long)N + n]);
    }
}

// ---------------------------------------------------------------------------
// Head split + RoPE + q-scale; pads DH 24 -> 32 with zeros.
// qp/kp layout: (H, L, 32) bf16
// ---------------------------------------------------------------------------
__global__ void esm_rope_split(const float* __restrict__ qt,
                               const float* __restrict__ kt,
                               __bf16* __restrict__ qp,
                               __bf16* __restrict__ kp) {
    int idx = blockIdx.x * blockDim.x + threadIdx.x;
    if (idx >= NHEAD * LSEQ * DHPAD) return;
    int d    = idx & (DHPAD - 1);
    int pos  = (idx >> 5) & (LSEQ - 1);
    int head = idx >> 14;
    if (d >= DHEAD) { qp[idx] = f2bf(0.f); kp[idx] = f2bf(0.f); return; }
    int j = (d < 12) ? d : d - 12;
    float inv = powf(10000.f, -(2.f * (float)j) / 24.f);
    float f = (float)pos * inv;
    float c = cosf(f), s = sinf(f);
    long base = (long)pos * DMODEL + head * DHEAD;
    float qv = qt[base + d], kv = kt[base + d];
    float qo, ko;
    if (d < 12) {
        qo = qv * c - qt[base + d + 12] * s;
        ko = kv * c - kt[base + d + 12] * s;
    } else {
        qo = qv * c + qt[base + d - 12] * s;
        ko = kv * c + kt[base + d - 12] * s;
    }
    qp[idx] = f2bf(qo * 0.20412414523193154f);  // 24^-0.5
    kp[idx] = f2bf(ko);
}

// V transpose per head: (L, D) f32 -> (H, 32, L) bf16, padded.
__global__ void esm_vt(const float* __restrict__ v, __bf16* __restrict__ vt) {
    int idx = blockIdx.x * blockDim.x + threadIdx.x;
    if (idx >= NHEAD * DHPAD * LSEQ) return;
    int pos  = idx & (LSEQ - 1);
    int d    = (idx >> 9) & (DHPAD - 1);
    int head = idx >> 14;
    vt[idx] = (d < DHEAD) ? f2bf(v[(long)pos * DMODEL + head * DHEAD + d])
                          : f2bf(0.f);
}

// ---------------------------------------------------------------------------
// Softmax: reads contiguous f32 scores from sbuf (H, L, L); writes bf16 probs
// (contiguous, for the P@V WMMA) and the final f32 probs into the strided
// d_out layout (q, k, layer*20+head) -- the only scattered pass.
// ---------------------------------------------------------------------------
__global__ void esm_softmax(const float* __restrict__ sbuf,
                            float* __restrict__ attn,
                            const int* __restrict__ amask,
                            __bf16* __restrict__ pbuf) {
    __shared__ float red[256];
    int q = blockIdx.x, head = blockIdx.y;
    const float* srow = sbuf + ((long)head * LSEQ + q) * LSEQ;
    float loc[2];
    float mx = -1e30f;
    for (int i = 0; i < 2; ++i) {
        int k = threadIdx.x + i * 256;
        float bias = (1.f - (float)amask[k]) * -1e9f;
        loc[i] = srow[k] + bias;
        mx = fmaxf(mx, loc[i]);
    }
    red[threadIdx.x] = mx; __syncthreads();
    for (int s = 128; s > 0; s >>= 1) {
        if (threadIdx.x < s) red[threadIdx.x] = fmaxf(red[threadIdx.x], red[threadIdx.x + s]);
        __syncthreads();
    }
    mx = red[0]; __syncthreads();
    float sum = 0.f;
    for (int i = 0; i < 2; ++i) { loc[i] = expf(loc[i] - mx); sum += loc[i]; }
    red[threadIdx.x] = sum; __syncthreads();
    for (int s = 128; s > 0; s >>= 1) {
        if (threadIdx.x < s) red[threadIdx.x] += red[threadIdx.x + s];
        __syncthreads();
    }
    float inv = 1.f / red[0];
    __bf16* prow = pbuf + ((long)head * LSEQ + q) * LSEQ;
    float* arow = attn + (long)q * (LSEQ * NLH) + head;
    for (int i = 0; i < 2; ++i) {
        int k = threadIdx.x + i * 256;
        float p = loc[i] * inv;
        prow[k] = f2bf(p);
        arow[(long)k * NLH] = p;   // single strided scatter into d_out
    }
}

// ---------------------------------------------------------------------------
// Generic bf16 WMMA GEMM:  C(M,N) = A(M,K) * Bt(N,K)^T  (+bias)(+residual)
// 256 threads = 8 waves; block tile 64x64; each wave owns TWO 16x16 M-tiles
// sharing one B fragment (two independent WMMA chains, 48B loads per WMMA).
// A fragment (ISA 16-bit A 16x32): lane L<16 holds row L, K in [0,8)+[16,24);
//   lane L>=16 holds row L-16, K in [8,16)+[24,32)  -> two v8bf loads.
// B fragment: lane holds Bt row (tileN + lane&15), same K split as A
//   ([0,8)+[16,24) or [8,16)+[24,32)) -> two v8bf loads.
// Output: out[cm*ldRowOut + cn*ldColOut + batch*strideOut]; epilogue modes:
//   0 = f32 (+bias,+residual), 1 = bf16, 2 = GELU(exact) -> bf16.
// Requires M % 64 == 0 (always 512 here).
// ---------------------------------------------------------------------------
__global__ void __launch_bounds__(256)
esm_wmma_gemm(const __bf16* __restrict__ A, const __bf16* __restrict__ B,
              const float* __restrict__ bias, const float* __restrict__ residual,
              float* __restrict__ outF, __bf16* __restrict__ outB,
              int M, int N, int K, int lda, int ldb,
              long ldRowOut, long ldColOut,
              long strideA, long strideB, long strideOut,
              int nValid, int mode) {
    int batch = blockIdx.z;
    A += strideA * batch;
    B += strideB * batch;
    long outOff = strideOut * batch;

    int wave = threadIdx.x >> 5;
    int lane = threadIdx.x & 31;
    int tileM0 = blockIdx.x * 64 + (wave >> 2) * 32;   // two M tiles: +0, +16
    int tileN  = blockIdx.y * 64 + (wave & 3) * 16;
    if (tileN >= N) return;  // wave-uniform exit, EXEC stays full

    int r  = lane & 15;
    int hi = lane >> 4;
    const __bf16* aPtr0 = A + (long)(tileM0 + r) * lda + hi * 8;
    const __bf16* aPtr1 = aPtr0 + (long)16 * lda;
    const __bf16* bPtr  = B + (long)(tileN + r) * ldb + hi * 8;

    v8f acc0 = {0.f, 0.f, 0.f, 0.f, 0.f, 0.f, 0.f, 0.f};
    v8f acc1 = {0.f, 0.f, 0.f, 0.f, 0.f, 0.f, 0.f, 0.f};
    for (int k = 0; k < K; k += 32) {
        if (k + 128 < K) {
            __builtin_prefetch(aPtr0 + k + 128, 0, 1);
            __builtin_prefetch(bPtr + k + 128, 0, 1);
        }
        union { v16bf v; v8bf h[2]; } a0, a1, b0;
        b0.h[0] = *(const v8bf*)(bPtr + k);
        b0.h[1] = *(const v8bf*)(bPtr + k + 16);
        v16bf bv = b0.v;
        a0.h[0] = *(const v8bf*)(aPtr0 + k);
        a0.h[1] = *(const v8bf*)(aPtr0 + k + 16);
        a1.h[0] = *(const v8bf*)(aPtr1 + k);
        a1.h[1] = *(const v8bf*)(aPtr1 + k + 16);
        acc0 = __builtin_amdgcn_wmma_f32_16x16x32_bf16(
                   false, a0.v, false, bv, (short)0, acc0, false, false);
        acc1 = __builtin_amdgcn_wmma_f32_16x16x32_bf16(
                   false, a1.v, false, bv, (short)0, acc1, false, false);
    }

    int cn = tileN + r;              // N index of this lane
    float bval = bias ? bias[cn] : 0.f;
    bool nOK = (cn < nValid);
    for (int t = 0; t < 2; ++t) {
        v8f acc = (t == 0) ? acc0 : acc1;
        int mBase = tileM0 + t * 16 + hi * 8;
        for (int rr = 0; rr < 8; ++rr) {
            int cm = mBase + rr;
            float v = acc[rr] + bval;
            if (residual) v += residual[(long)cm * N + cn];
            if (nOK) {
                long o = (long)cm * ldRowOut + (long)cn * ldColOut + outOff;
                if (mode == 0) {
                    outF[o] = v;
                } else if (mode == 1) {
                    outB[o] = f2bf(v);
                } else {
                    float g = 0.5f * v * (1.f + erff(v * 0.70710678118654752f));
                    outB[o] = f2bf(g);
                }
            }
        }
    }
}

// ---------------------------------------------------------------------------
extern "C" void kernel_launch(void* const* d_in, const int* in_sizes, int n_in,
                              void* d_out, int out_size, void* d_ws, size_t ws_size,
                              hipStream_t stream) {
    const int*   ids   = (const int*)d_in[0];
    const int*   amask = (const int*)d_in[1];
    const float* we    = (const float*)d_in[2];
    const float* ln1w  = (const float*)d_in[3];
    const float* ln1b  = (const float*)d_in[4];
    const float* Wq    = (const float*)d_in[5];
    const float* bq    = (const float*)d_in[6];
    const float* Wk    = (const float*)d_in[7];
    const float* bk    = (const float*)d_in[8];
    const float* Wv    = (const float*)d_in[9];
    const float* bv    = (const float*)d_in[10];
    const float* Wo    = (const float*)d_in[11];
    const float* bo    = (const float*)d_in[12];
    const float* ln2w  = (const float*)d_in[13];
    const float* ln2b  = (const float*)d_in[14];
    const float* Wi    = (const float*)d_in[15];
    const float* bi    = (const float*)d_in[16];
    const float* Wo2   = (const float*)d_in[17];
    const float* bo2   = (const float*)d_in[18];
    const float* lnfw  = (const float*)d_in[19];
    const float* lnfb  = (const float*)d_in[20];

    float* outRep  = (float*)d_out;
    float* outAttn = outRep + (long)LSEQ * DMODEL;

    // Workspace carve-out (256B aligned)
    char* wp = (char*)d_ws;
    auto carve = [&](size_t bytes) {
        char* p = wp;
        wp += (bytes + 255) & ~(size_t)255;
        return p;
    };
    float*  scale = (float*)carve(sizeof(float));
    float*  h     = (float*)carve((size_t)LSEQ * DMODEL * 4);
    float*  h1    = (float*)carve((size_t)LSEQ * DMODEL * 4);
    __bf16* xb    = (__bf16*)carve((size_t)LSEQ * DMODEL * 2);
    float*  qt    = (float*)carve((size_t)LSEQ * DMODEL * 4);
    float*  kt    = (float*)carve((size_t)LSEQ * DMODEL * 4);
    float*  vtF   = (float*)carve((size_t)LSEQ * DMODEL * 4);
    __bf16* qp    = (__bf16*)carve((size_t)NHEAD * LSEQ * DHPAD * 2);
    __bf16* kp    = (__bf16*)carve((size_t)NHEAD * LSEQ * DHPAD * 2);
    __bf16* vtp   = (__bf16*)carve((size_t)NHEAD * DHPAD * LSEQ * 2);
    float*  sbuf  = (float*)carve((size_t)NHEAD * LSEQ * LSEQ * 4);
    __bf16* pbuf  = (__bf16*)carve((size_t)NHEAD * LSEQ * LSEQ * 2);
    __bf16* ctxb  = (__bf16*)carve((size_t)LSEQ * DMODEL * 2);
    __bf16* gbuf  = (__bf16*)carve((size_t)LSEQ * FDIM * 2);
    __bf16* WqT   = (__bf16*)carve((size_t)DMODEL * DMODEL * 2);
    __bf16* WkT   = (__bf16*)carve((size_t)DMODEL * DMODEL * 2);
    __bf16* WvT   = (__bf16*)carve((size_t)DMODEL * DMODEL * 2);
    __bf16* WoT   = (__bf16*)carve((size_t)DMODEL * DMODEL * 2);
    __bf16* WiT   = (__bf16*)carve((size_t)FDIM * DMODEL * 2);
    __bf16* Wo2T  = (__bf16*)carve((size_t)DMODEL * FDIM * 2);

    esm_mask_stats<<<1, 512, 0, stream>>>(ids, amask, scale);
    esm_embed<<<LSEQ, 256, 0, stream>>>(ids, amask, we, scale, h);

    dim3 blk(256);
    dim3 gDD(LSEQ / 64, (DMODEL + 63) / 64, 1);    // 480-wide GEMMs
    int  gtDD = (DMODEL * DMODEL + 255) / 256;
    int  gtDF = (DMODEL * FDIM + 255) / 256;
    int  nSplit = NHEAD * LSEQ * DHPAD;

    for (int layer = 0; layer < NLAYER; ++layer) {
        long oDD = (long)layer * DMODEL * DMODEL;
        long oDF = (long)layer * DMODEL * FDIM;
        const float* l1w = ln1w + layer * DMODEL;
        const float* l1b = ln1b + layer * DMODEL;
        const float* l2w = ln2w + layer * DMODEL;
        const float* l2b = ln2b + layer * DMODEL;

        // weights -> transposed bf16
        esm_w2bf_t<<<gtDD, blk, 0, stream>>>(Wq + oDD, WqT, DMODEL, DMODEL);
        esm_w2bf_t<<<gtDD, blk, 0, stream>>>(Wk + oDD, WkT, DMODEL, DMODEL);
        esm_w2bf_t<<<gtDD, blk, 0, stream>>>(Wv + oDD, WvT, DMODEL, DMODEL);
        esm_w2bf_t<<<gtDD, blk, 0, stream>>>(Wo + oDD, WoT, DMODEL, DMODEL);
        esm_w2bf_t<<<gtDF, blk, 0, stream>>>(Wi + oDF, WiT, DMODEL, FDIM);
        esm_w2bf_t<<<gtDF, blk, 0, stream>>>(Wo2 + oDF, Wo2T, FDIM, DMODEL);

        // LN1 -> xb (bf16)
        esm_layernorm<<<LSEQ, 256, 0, stream>>>(h, l1w, l1b, nullptr, xb);

        // QKV projections (f32 out)
        esm_wmma_gemm<<<gDD, 256, 0, stream>>>(xb, WqT, bq + layer * DMODEL, nullptr,
            qt, nullptr, LSEQ, DMODEL, DMODEL, DMODEL, DMODEL,
            DMODEL, 1, 0, 0, 0, DMODEL, 0);
        esm_wmma_gemm<<<gDD, 256, 0, stream>>>(xb, WkT, bk + layer * DMODEL, nullptr,
            kt, nullptr, LSEQ, DMODEL, DMODEL, DMODEL, DMODEL,
            DMODEL, 1, 0, 0, 0, DMODEL, 0);
        esm_wmma_gemm<<<gDD, 256, 0, stream>>>(xb, WvT, bv + layer * DMODEL, nullptr,
            vtF, nullptr, LSEQ, DMODEL, DMODEL, DMODEL, DMODEL,
            DMODEL, 1, 0, 0, 0, DMODEL, 0);

        // RoPE + head split (padded), V transpose
        esm_rope_split<<<(nSplit + 255) / 256, blk, 0, stream>>>(qt, kt, qp, kp);
        esm_vt<<<(nSplit + 255) / 256, blk, 0, stream>>>(vtF, vtp);

        // scores = q @ k^T -> contiguous f32 sbuf (H, L, L)
        dim3 gS(LSEQ / 64, LSEQ / 64, NHEAD);
        esm_wmma_gemm<<<gS, 256, 0, stream>>>(qp, kp, nullptr, nullptr,
            sbuf, nullptr, LSEQ, LSEQ, DHPAD, DHPAD, DHPAD,
            LSEQ, 1,
            (long)LSEQ * DHPAD, (long)LSEQ * DHPAD, (long)LSEQ * LSEQ, LSEQ, 0);

        // softmax: contiguous read, bf16 probs + single strided scatter to d_out
        float* attnBase = outAttn + layer * NHEAD;
        dim3 gSm(LSEQ, NHEAD);
        esm_softmax<<<gSm, 256, 0, stream>>>(sbuf, attnBase, amask, pbuf);

        // ctx = probs @ V  -> ctxb (L, 480) bf16, only 24 of 32 cols valid
        dim3 gC(LSEQ / 64, 1, NHEAD);
        esm_wmma_gemm<<<gC, 256, 0, stream>>>(pbuf, vtp, nullptr, nullptr,
            nullptr, ctxb, LSEQ, DHPAD, LSEQ, LSEQ, LSEQ,
            DMODEL, 1,
            (long)LSEQ * LSEQ, (long)DHPAD * LSEQ, DHEAD, DHEAD, 1);

        // attention out projection + residual -> h1
        esm_wmma_gemm<<<gDD, 256, 0, stream>>>(ctxb, WoT, bo + layer * DMODEL, h,
            h1, nullptr, LSEQ, DMODEL, DMODEL, DMODEL, DMODEL,
            DMODEL, 1, 0, 0, 0, DMODEL, 0);

        // LN2 -> xb
        esm_layernorm<<<LSEQ, 256, 0, stream>>>(h1, l2w, l2b, nullptr, xb);

        // FFN up + GELU -> gbuf bf16
        dim3 gF1(LSEQ / 64, FDIM / 64, 1);
        esm_wmma_gemm<<<gF1, 256, 0, stream>>>(xb, WiT, bi + layer * FDIM, nullptr,
            nullptr, gbuf, LSEQ, FDIM, DMODEL, DMODEL, DMODEL,
            FDIM, 1, 0, 0, 0, FDIM, 2);

        // FFN down + residual -> h (next layer input)
        esm_wmma_gemm<<<gDD, 256, 0, stream>>>(gbuf, Wo2T, bo2 + layer * DMODEL, h1,
            h, nullptr, LSEQ, DMODEL, FDIM, FDIM, FDIM,
            DMODEL, 1, 0, 0, 0, DMODEL, 0);
    }

    // final LN -> representations (f32 in d_out)
    esm_layernorm<<<LSEQ, 256, 0, stream>>>(h, lnfw, lnfb, outRep, nullptr);
}
